// EncoderDecoderLSTM_80650895884783
// MI455X (gfx1250) — matrix operand order repaired and checked by reading, not compiled, not run on hardware
//
#include <hip/hip_runtime.h>
#include <hip/hip_bf16.h>

// ---------------------------------------------------------------------------
// EncoderDecoderLSTM for MI455X (gfx1250), bf16 WMMA with fp32 accumulation.
//
//   1. convert input + transpose-convert all weights to bf16 (N x K layout)
//   2. enc0 = in @ W0 + b0 ; enc = enc0 @ W1 + b1      (2x4-blocked WMMA GEMM)
//   3. xg_lastc = enc[:, -1, :] @ Wih + bih + bhh      (fp32, 2MB)
//   4. 640 sequential step kernels: WG = 8 waves (8 m-tiles) sharing one jt;
//      B tiles staged in LDS via async global->LDS, double-buffered in
//      4-chunk super-stages; fused LSTM cell on the accumulators.
//   5. out = (hs @ dec_W0 + b0) @ dec_W1 + b1          (2x4-blocked WMMA GEMM)
// ---------------------------------------------------------------------------

typedef __attribute__((ext_vector_type(16))) __bf16 v16bf;
typedef __attribute__((ext_vector_type(8)))  __bf16 v8bf;
typedef __attribute__((ext_vector_type(8)))  float  v8f;
typedef __attribute__((ext_vector_type(4)))  int    v4i;

#define B_N     128
#define PAST_N  512
#define F_IN_N  64
#define ENC_HN  128
#define IP_N    512
#define H_N     1024
#define DEC_HN  512
#define F_OUT_N 64
#define FUT_N   128
#define TOT_T   (PAST_N + FUT_N)   /* 640 */
#define G4      (4 * H_N)          /* 4096 */

// ---------------- async global -> LDS copy (CDNA5 path w/ fallback) --------
// clang's builtin takes (int4 AS1* src, int4 AS3* dst, imm offset, imm cpol).
typedef __attribute__((address_space(1))) v4i gv4i;
typedef __attribute__((address_space(3))) v4i lv4i;

#if __has_builtin(__builtin_amdgcn_global_load_async_to_lds_b128)
#define HAVE_ASYNC_LDS 1
#endif

__device__ __forceinline__ void async_copy16(const __bf16* g, __bf16* l) {
#if defined(HAVE_ASYNC_LDS)
  __builtin_amdgcn_global_load_async_to_lds_b128((gv4i*)g, (lv4i*)l, 0, 0);
#else
  *(v8bf*)l = *(const v8bf*)g;
#endif
}

__device__ __forceinline__ void wait_async_lds() {
#if defined(HAVE_ASYNC_LDS)
#if __has_builtin(__builtin_amdgcn_s_wait_asynccnt)
  __builtin_amdgcn_s_wait_asynccnt(0);
#else
  asm volatile("s_wait_asynccnt 0x0" ::: "memory");
#endif
#endif
}

// ---- WMMA tile loaders (wave32; blockDim.x multiple of 32, all lanes on) ----
// A matrix (16x32, row-major M x K source):
//   lanes 0-15: row M=lane,    K = k0 + 0..15 (contiguous)
//   lanes16-31: row M=lane-16, K = k0 + 16..31
__device__ __forceinline__ v16bf load_tile_A(const __bf16* base, int row0,
                                             size_t ld, int k0) {
  int lane = threadIdx.x & 31;
  int r    = row0 + (lane & 15);
  int off  = (lane >> 4) << 4;
  const __bf16* p = base + (size_t)r * ld + (size_t)(k0 + off);
  v8bf lo = *(const v8bf*)(p);
  v8bf hi = *(const v8bf*)(p + 8);
  v16bf out;
#pragma unroll
  for (int i = 0; i < 8; ++i) { out[i] = lo[i]; out[i + 8] = hi[i]; }
  return out;
}

// B matrix (32x16 K x N), sourced from transposed weights stored N x K:
//   lanes 0-15: col N=lane,    K = k0 + 0..15 (contiguous)
//   lanes16-31: col N=lane-16, K = k0 + 16..31
__device__ __forceinline__ v16bf load_tile_B(const __bf16* base, int row0,
                                             size_t ld, int k0) {
  int lane = threadIdx.x & 31;
  int r    = row0 + (lane & 15);
  int kb   = k0 + ((lane >> 4) << 4);
  const __bf16* p = base + (size_t)r * ld + (size_t)kb;
  v8bf lo = *(const v8bf*)(p);
  v8bf hi = *(const v8bf*)(p + 8);
  v16bf out;
#pragma unroll
  for (int i = 0; i < 8; ++i) { out[i] = lo[i]; out[i + 8] = hi[i]; }
  return out;
}

__device__ __forceinline__ v8f wmma_bf16(v16bf a, v16bf b, v8f c) {
  return __builtin_amdgcn_wmma_f32_16x16x32_bf16(false, a, false, b,
                                                 (short)0, c, false, false);
}

__device__ __forceinline__ float sigmoidf_(float x) {
  return 1.0f / (1.0f + __expf(-x));
}

// ---- 2x4 register-blocked GEMM: C = A(MxK) * BT(NxK)^T + bias --------------
// One wave per 32x64 output block (8 WMMA tiles): 8 WMMAs per 6 tile loads,
// ~130 VGPRs live -> no accumulator spilling.
__global__ void __launch_bounds__(256, 1)
gemm_bf16_kernel(const __bf16* __restrict__ A, size_t lda,
                 const __bf16* __restrict__ BT, int K,
                 const float* __restrict__ bias,
                 float* __restrict__ Cf,
                 __bf16* __restrict__ Cb, size_t ldc,
                 int Mg, int Ng) {   // groups: Mg = Mtiles/2, Ng = Ntiles/4
  int wave  = (int)((blockIdx.x * blockDim.x + threadIdx.x) >> 5);
  int nwave = (int)((gridDim.x * blockDim.x) >> 5);
  int lane  = threadIdx.x & 31;
  int total = Mg * Ng;
  for (int t = wave; t < total; t += nwave) {
    int mg = t / Ng;
    int ng = t - mg * Ng;
    v8f acc[2][4];
#pragma unroll
    for (int i = 0; i < 2; ++i)
#pragma unroll
      for (int j = 0; j < 4; ++j)
        acc[i][j] = (v8f){0.f, 0.f, 0.f, 0.f, 0.f, 0.f, 0.f, 0.f};
    for (int k0 = 0; k0 < K; k0 += 32) {
      v16bf a[2], b[4];
#pragma unroll
      for (int i = 0; i < 2; ++i)
        a[i] = load_tile_A(A, (mg * 2 + i) * 16, lda, k0);
#pragma unroll
      for (int j = 0; j < 4; ++j)
        b[j] = load_tile_B(BT, (ng * 4 + j) * 16, (size_t)K, k0);
#pragma unroll
      for (int i = 0; i < 2; ++i)
#pragma unroll
        for (int j = 0; j < 4; ++j)
          acc[i][j] = wmma_bf16(a[i], b[j], acc[i][j]);
    }
#pragma unroll
    for (int i = 0; i < 2; ++i)
#pragma unroll
      for (int j = 0; j < 4; ++j) {
        int n     = (ng * 4 + j) * 16 + (lane & 15);
        int mbase = (mg * 2 + i) * 16 + ((lane >> 4) << 3);
        float bv  = bias ? bias[n] : 0.0f;
#pragma unroll
        for (int v = 0; v < 8; ++v) {
          float val  = acc[i][j][v] + bv;
          size_t idx = (size_t)(mbase + v) * ldc + (size_t)n;
          if (Cf) Cf[idx] = val;
          else    Cb[idx] = (__bf16)val;
        }
      }
  }
}

// ---- One LSTM timestep. WG = 256 threads = 8 waves = 8 m-tiles, one jt. ----
// B tiles (4 gates x 4 k-chunks = 16 KB) staged in LDS, double-buffered.
__global__ void lstm_step_kernel(const __bf16* __restrict__ encT,  // t-th slice, ld = PAST_N*IP_N; null for future
                                 const __bf16* __restrict__ WihT,  // 4096 x 512
                                 const __bf16* __restrict__ hprev, size_t ld_h,
                                 const __bf16* __restrict__ WhhT,  // 4096 x 1024
                                 const float* __restrict__ addvec, int addstride,
                                 float* __restrict__ cbuf,
                                 __bf16* __restrict__ hout, size_t ld_hout) {
  __shared__ alignas(16) __bf16 sB[2][8192];   // 2 x 16KB: [cl(4)][gate(4)][16][32]
  const int tid  = threadIdx.x;
  const int wave = tid >> 5;        // 0..7  -> mt
  const int lane = tid & 31;
  const int mt   = wave;
  const int jt   = blockIdx.x;      // 0..63

  const int encChunks = encT ? (IP_N / 32) : 0;        // 16 or 0
  const int nstages   = (encChunks + H_N / 32) / 4;    // 12 or 8

  // stage one super-stage (4 k-chunks of B: 8192 bf16) into sB[buf]
  auto stageB = [&](int stg, int buf) {
    const __bf16* Bb;
    size_t ldb;
    int kb;
    int cg = stg * 4;
    if (cg < encChunks) { Bb = WihT; ldb = IP_N; kb = cg * 32; }
    else                { Bb = WhhT; ldb = H_N;  kb = (cg - encChunks) * 32; }
    int rem = tid * 8;                       // 0..2047, per-chunk element idx
    int g   = rem >> 9;                      // gate 0..3
    int rr  = (rem >> 5) & 15;               // row in tile
    int c0  = rem & 31;                      // k offset in chunk
    const __bf16* gsrc = Bb + (size_t)(g * H_N + jt * 16 + rr) * ldb;
#pragma unroll
    for (int cl = 0; cl < 4; ++cl)
      async_copy16(gsrc + (size_t)(kb + cl * 32 + c0),
                   &sB[buf][cl * 2048 + rem]);
  };

  v8f a0 = {0.f, 0.f, 0.f, 0.f, 0.f, 0.f, 0.f, 0.f};
  v8f a1 = a0, a2 = a0, a3 = a0;

  stageB(0, 0);
  wait_async_lds();
  __syncthreads();

  for (int s = 0; s < nstages; ++s) {
    int buf = s & 1;
    if (s + 1 < nstages) stageB(s + 1, buf ^ 1);
    // A source for this super-stage (stage boundaries align with enc->h switch)
    const __bf16* Ab;
    size_t la;
    int ka;
    if (s * 4 < encChunks) { Ab = encT;  la = (size_t)PAST_N * IP_N; ka = s * 4 * 32; }
    else                   { Ab = hprev; la = ld_h;  ka = (s * 4 - encChunks) * 32; }
#pragma unroll
    for (int cl = 0; cl < 4; ++cl) {
      v16bf a = load_tile_A(Ab, mt * 16, la, ka + cl * 32);
      const __bf16* bb = &sB[buf][cl * 2048];
      a0 = wmma_bf16(a, load_tile_B(bb + 0 * 512, 0, 32, 0), a0);
      a1 = wmma_bf16(a, load_tile_B(bb + 1 * 512, 0, 32, 0), a1);
      a2 = wmma_bf16(a, load_tile_B(bb + 2 * 512, 0, 32, 0), a2);
      a3 = wmma_bf16(a, load_tile_B(bb + 3 * 512, 0, 32, 0), a3);
    }
    wait_async_lds();
    __syncthreads();
  }

  // fused LSTM cell on the four gate accumulators
  int j     = jt * 16 + (lane & 15);
  int mbase = mt * 16 + ((lane >> 4) << 3);
#pragma unroll
  for (int v = 0; v < 8; ++v) {
    int m = mbase + v;
    const float* av = addvec + (size_t)m * (size_t)addstride;
    float gi = a0[v] + av[j];
    float gf = a1[v] + av[H_N + j];
    float gg = a2[v] + av[2 * H_N + j];
    float go = a3[v] + av[3 * H_N + j];
    float si = sigmoidf_(gi), sf = sigmoidf_(gf), so = sigmoidf_(go);
    float tg = tanhf(gg);
    size_t ci = (size_t)m * H_N + (size_t)j;
    float c = sf * cbuf[ci] + si * tg;
    cbuf[ci] = c;
    hout[(size_t)m * ld_hout + (size_t)j] = (__bf16)(so * tanhf(c));
  }
}

// ---- small utility kernels ----
__global__ void f32_to_bf16_kernel(const float* __restrict__ s,
                                   __bf16* __restrict__ d, int n) {
  int i = blockIdx.x * blockDim.x + threadIdx.x;
  if (i < n) d[i] = (__bf16)s[i];
}
__global__ void transpose_to_bf16_kernel(const float* __restrict__ W,  // K x N
                                         __bf16* __restrict__ WT,      // N x K
                                         int K, int N) {
  int i = blockIdx.x * blockDim.x + threadIdx.x;
  if (i < K * N) {
    int k = i / N, n = i - k * N;
    WT[(size_t)n * K + k] = (__bf16)W[i];
  }
}
__global__ void combine_bias_kernel(const float* __restrict__ a,
                                    const float* __restrict__ b,
                                    float* __restrict__ o, int n) {
  int i = blockIdx.x * blockDim.x + threadIdx.x;
  if (i < n) o[i] = a[i] + b[i];
}
__global__ void zero_state_kernel(__bf16* __restrict__ h0,
                                  float* __restrict__ c0, int n) {
  int i = blockIdx.x * blockDim.x + threadIdx.x;
  if (i < n) { h0[i] = (__bf16)0.0f; c0[i] = 0.0f; }
}

// ---- host-side launcher ----
static inline void launch_gemm(const __bf16* A, size_t lda, const __bf16* BT,
                               int K, const float* bias, float* Cf, __bf16* Cb,
                               size_t ldc, int Mtiles, int Ntiles,
                               hipStream_t s) {
  int Mg = Mtiles / 2, Ng = Ntiles / 4;   // all shapes divide by 2 / 4
  int groups = Mg * Ng;
  int blocks = (groups + 7) / 8;          // 8 waves (256 threads) per block
  if (blocks > 4096) blocks = 4096;
  gemm_bf16_kernel<<<blocks, 256, 0, s>>>(A, lda, BT, K, bias, Cf, Cb, ldc,
                                          Mg, Ng);
}

extern "C" void kernel_launch(void* const* d_in, const int* in_sizes, int n_in,
                              void* d_out, int out_size, void* d_ws,
                              size_t ws_size, hipStream_t stream) {
  const float* input_seq = (const float*)d_in[0];
  const float* enc_W0    = (const float*)d_in[1];
  const float* enc_b0    = (const float*)d_in[2];
  const float* enc_W1    = (const float*)d_in[3];
  const float* enc_b1    = (const float*)d_in[4];
  const float* Wih       = (const float*)d_in[5];
  const float* bih       = (const float*)d_in[6];
  const float* Whh       = (const float*)d_in[7];
  const float* bhh       = (const float*)d_in[8];
  const float* dec_W0    = (const float*)d_in[9];
  const float* dec_b0    = (const float*)d_in[10];
  const float* dec_W1    = (const float*)d_in[11];
  const float* dec_b1    = (const float*)d_in[12];
  float* out = (float*)d_out;

  // ---- workspace layout ----
  char* ws = (char*)d_ws;
  size_t off = 0;
  auto alloc = [&](size_t bytes) -> char* {
    char* p = ws + off;
    off += (bytes + 255) & ~(size_t)255;
    return p;
  };
  __bf16* in_bf   = (__bf16*)alloc((size_t)B_N * PAST_N * F_IN_N * 2);  //  8 MB
  __bf16* enc0    = (__bf16*)alloc((size_t)B_N * PAST_N * ENC_HN * 2);  // 16 MB
  __bf16* enc     = (__bf16*)alloc((size_t)B_N * PAST_N * IP_N * 2);    // 64 MB
  __bf16* encW0T  = (__bf16*)alloc((size_t)ENC_HN * F_IN_N * 2);
  __bf16* encW1T  = (__bf16*)alloc((size_t)IP_N * ENC_HN * 2);
  __bf16* WihT    = (__bf16*)alloc((size_t)G4 * IP_N * 2);              //  4 MB
  __bf16* WhhT    = (__bf16*)alloc((size_t)G4 * H_N * 2);               //  8 MB
  __bf16* decW0T  = (__bf16*)alloc((size_t)DEC_HN * H_N * 2);           //  1 MB
  __bf16* decW1T  = (__bf16*)alloc((size_t)F_OUT_N * DEC_HN * 2);
  float*  bihbhh  = (float*)alloc((size_t)G4 * 4);
  float*  xglastc = (float*)alloc((size_t)B_N * G4 * 4);                //  2 MB
  __bf16* zerobuf = (__bf16*)alloc((size_t)B_N * H_N * 2);
  float*  cbuf    = (float*)alloc((size_t)B_N * H_N * 4);
  __bf16* hs      = (__bf16*)alloc((size_t)B_N * TOT_T * H_N * 2);      // 160 MB
  // decoder intermediate reuses the encoder region (dead after the LSTM loop)
  __bf16* dint    = (__bf16*)ws;                                        // 80 MB

  // ---- phase 1: convert / transpose ----
  int nIn = B_N * PAST_N * F_IN_N;
  f32_to_bf16_kernel<<<(nIn + 255) / 256, 256, 0, stream>>>(input_seq, in_bf, nIn);
  transpose_to_bf16_kernel<<<(F_IN_N * ENC_HN + 255) / 256, 256, 0, stream>>>(enc_W0, encW0T, F_IN_N, ENC_HN);
  transpose_to_bf16_kernel<<<(ENC_HN * IP_N + 255) / 256, 256, 0, stream>>>(enc_W1, encW1T, ENC_HN, IP_N);
  transpose_to_bf16_kernel<<<(IP_N * G4 + 255) / 256, 256, 0, stream>>>(Wih, WihT, IP_N, G4);
  transpose_to_bf16_kernel<<<(H_N * G4 + 255) / 256, 256, 0, stream>>>(Whh, WhhT, H_N, G4);
  transpose_to_bf16_kernel<<<(H_N * DEC_HN + 255) / 256, 256, 0, stream>>>(dec_W0, decW0T, H_N, DEC_HN);
  transpose_to_bf16_kernel<<<(DEC_HN * F_OUT_N + 255) / 256, 256, 0, stream>>>(dec_W1, decW1T, DEC_HN, F_OUT_N);
  combine_bias_kernel<<<(G4 + 255) / 256, 256, 0, stream>>>(bih, bhh, bihbhh, G4);
  zero_state_kernel<<<(B_N * H_N + 255) / 256, 256, 0, stream>>>(zerobuf, cbuf, B_N * H_N);

  // ---- phase 2: encoder ----
  launch_gemm(in_bf, F_IN_N, encW0T, F_IN_N, enc_b0, nullptr, enc0, ENC_HN,
              (B_N * PAST_N) / 16, ENC_HN / 16, stream);
  launch_gemm(enc0, ENC_HN, encW1T, ENC_HN, enc_b1, nullptr, enc, IP_N,
              (B_N * PAST_N) / 16, IP_N / 16, stream);

  // ---- phase 3: xg_lastc = enc[:, -1, :] @ Wih + (bih + bhh) ----
  launch_gemm(enc + (size_t)(PAST_N - 1) * IP_N, (size_t)PAST_N * IP_N, WihT,
              IP_N, bihbhh, xglastc, nullptr, G4, B_N / 16, G4 / 16, stream);

  // ---- phase 4: sequential LSTM (640 steps; 64 WGs x 8 waves each) ----
  const size_t ld_hs = (size_t)TOT_T * H_N;
  for (int t = 0; t < PAST_N; ++t) {
    const __bf16* hp = (t == 0) ? zerobuf : (hs + (size_t)(t - 1) * H_N);
    size_t ld_h      = (t == 0) ? (size_t)H_N : ld_hs;
    lstm_step_kernel<<<64, 256, 0, stream>>>(
        enc + (size_t)t * IP_N, WihT, hp, ld_h, WhhT, bihbhh, /*addstride=*/0,
        cbuf, hs + (size_t)t * H_N, ld_hs);
  }
  for (int t = PAST_N; t < TOT_T; ++t) {
    lstm_step_kernel<<<64, 256, 0, stream>>>(
        nullptr, nullptr, hs + (size_t)(t - 1) * H_N, ld_hs, WhhT, xglastc,
        /*addstride=*/G4, cbuf, hs + (size_t)t * H_N, ld_hs);
  }

  // ---- phase 5: decoder over all 81920 hidden rows ----
  launch_gemm(hs, H_N, decW0T, H_N, dec_b0, nullptr, dint, DEC_HN,
              (B_N * TOT_T) / 16, DEC_HN / 16, stream);
  launch_gemm(dint, DEC_HN, decW1T, DEC_HN, dec_b1, out, nullptr, F_OUT_N,
              (B_N * TOT_T) / 16, F_OUT_N / 16, stream);
}
